// GFR2MambaMixer_46351287058505
// MI455X (gfx1250) — matrix operand order repaired and checked by reading, not compiled, not run on hardware
//
#include <hip/hip_runtime.h>
#include <hip/hip_bf16.h>

// ---------------- problem constants (from reference) ----------------
#define B2      2
#define LSEQ    2048
#define DMODEL  1024
#define NH      32        // H
#define PDIM    64        // P
#define NDIM    128       // N
#define KCONV   4
#define CHUNK2  256
#define NCH     8         // LSEQ / CHUNK
#define INTER2  2048      // H*P
#define CONVD   2304      // INTER + 2*G*N
#define PROJ2   4384      // INTER + CONVD + H
#define MTOK    (B2*LSEQ) // 4096

typedef __bf16 bf16;
typedef __attribute__((ext_vector_type(8)))  __bf16 v8bf;
typedef __attribute__((ext_vector_type(16))) __bf16 v16bf;
typedef __attribute__((ext_vector_type(8)))  float  v8f;
typedef __attribute__((ext_vector_type(4)))  unsigned v4u;
typedef __attribute__((ext_vector_type(8)))  int      v8i;
typedef __attribute__((ext_vector_type(4)))  int      v4i;

union Frag16 { v16bf v; v8bf h[2]; };

__device__ inline v8f zero_v8f() {
  v8f z = {0.f, 0.f, 0.f, 0.f, 0.f, 0.f, 0.f, 0.f};
  return z;
}

// ---------------- TDM availability ----------------
#if defined(__has_builtin)
#if __has_builtin(__builtin_amdgcn_tensor_load_to_lds) && __has_builtin(__builtin_amdgcn_s_wait_tensorcnt)
#define USE_TDM 1
#endif
#endif
#ifndef USE_TDM
#define USE_TDM 0
#endif

#if USE_TDM
// Issue one TDM 2D tile load: rows x 32 bf16 elements, row stride = rowLen elems.
// D# packing per CDNA5 ISA 8.3/8.4 (group0: count/lds/global/type, group1: dims).
// Toolchain here exposes the 6-arg builtin: (v4u, v8i, v4i, v4i, v8i, i32 cpol).
__device__ inline void tdm_load_tile_bf16(const bf16* gtile, unsigned ldsOff,
                                          int rows, int rowLenElems, int totRows) {
  unsigned long long ga = (unsigned long long)(size_t)gtile;
  v4u g0;
  g0.x = 1u;                                                  // count=1, user mode
  g0.y = ldsOff;                                              // lds_addr [63:32]
  g0.z = (unsigned)(ga & 0xffffffffu);                        // global_addr [95:64]
  g0.w = (unsigned)((ga >> 32) & 0x01ffffffu) | 0x80000000u;  // addr hi + type=2
  v8i g1;
  g1[0] = (int)(1u << 16);                                    // data_size=1 (2B), mask=0
  g1[1] = (int)(((unsigned)rowLenElems & 0xffffu) << 16);     // tensor_dim0 lo16 @48
  g1[2] = (int)((((unsigned)rowLenElems >> 16) & 0xffffu) |
                (((unsigned)totRows & 0xffffu) << 16));       // dim0 hi16 | dim1 lo16
  g1[3] = (int)((((unsigned)totRows >> 16) & 0xffffu) |
                (32u << 16));                                 // dim1 hi16 | tile_dim0=32
  g1[4] = (int)((unsigned)rows & 0xffffu);                    // tile_dim1=rows, tile_dim2=0
  g1[5] = (int)(unsigned)rowLenElems;                         // dim0_stride lo32
  g1[6] = 0;                                                  // dim0_stride hi, dim1_stride lo
  g1[7] = 0;
  v4i g2 = {0, 0, 0, 0};
  v4i g3 = {0, 0, 0, 0};
  v8i g4 = {0, 0, 0, 0, 0, 0, 0, 0};
  __builtin_amdgcn_tensor_load_to_lds(g0, g1, g2, g3, g4, 0);
}
#endif

// ---------------- fp32 -> bf16 convert ----------------
__global__ __launch_bounds__(256) void cvt_bf16_kernel(const float* __restrict__ src,
                                                       bf16* __restrict__ dst, int n) {
  int i = blockIdx.x * 256 + threadIdx.x;
  if (i < n) dst[i] = (bf16)src[i];
}

// ---------------- WMMA bf16 GEMM:  C[M,N] = A[M,K] * Bw[N,K]^T ----------------
// block = 256 threads = 8 waves; block tile 256(M) x NT*16(N); wave tile 32 x NT*16.
// A tile (256 x 32 bf16) staged in LDS via TDM (double-buffered); B read from L2.
template <int NT>
__global__ __launch_bounds__(256) void gemm_bf16_lds(const bf16* __restrict__ A,
                                                     const bf16* __restrict__ Bw,
                                                     float* __restrict__ C,
                                                     int M, int N, int K, int n_base) {
  const int lane = threadIdx.x & 31;
  const int wave = threadIdx.x >> 5;
  const int mBlk = blockIdx.x * 256;
  const int n0 = n_base + blockIdx.y * (NT * 16);

  __shared__ bf16 sA[2][256 * 32];

  // fragment addressing per ISA VGPR layouts (wave32)
  const int arow = lane & 15;            // A: lanes 0..15 / 16..31 both hold M=0..15
  const int akb  = (lane >> 4) << 3;     // A: k chunks at kb and kb+16
  const int bcol = lane & 15;            // B: N = lane&15
  const int bkb  = (lane >> 4) << 3;     // B: k chunks at kb and kb+16

  const bf16* Bp[NT];
#pragma unroll
  for (int t = 0; t < NT; ++t)
    Bp[t] = Bw + (size_t)(n0 + 16 * t + bcol) * K + bkb;

  v8f acc[2][NT];
#pragma unroll
  for (int s = 0; s < 2; ++s)
#pragma unroll
    for (int t = 0; t < NT; ++t) acc[s][t] = zero_v8f();

  // ---- preload first A tile into buffer 0 ----
#if USE_TDM
  if (wave == 0) {
    tdm_load_tile_bf16(A + (size_t)mBlk * K, (unsigned)(size_t)&sA[0][0], 256, K, M);
    __builtin_amdgcn_s_wait_tensorcnt(0);
  }
  __syncthreads();
#else
  for (int idx = threadIdx.x; idx < 1024; idx += 256) {
    int row = idx >> 2, cc = (idx & 3) << 3;
    *reinterpret_cast<v8bf*>(&sA[0][row * 32 + cc]) =
        *reinterpret_cast<const v8bf*>(A + (size_t)(mBlk + row) * K + cc);
  }
  __syncthreads();
#endif

  int buf = 0;
  for (int k0 = 0; k0 < K; k0 += 32) {
    const bool more = (k0 + 32) < K;
#if USE_TDM
    if (wave == 0 && more)
      tdm_load_tile_bf16(A + (size_t)mBlk * K + (k0 + 32),
                         (unsigned)(size_t)&sA[buf ^ 1][0], 256, K, M);
#endif
    if (k0 + 128 < K) __builtin_prefetch(Bp[0] + k0 + 128, 0, 1);

    // B fragments (shared across both M-subtiles)
    Frag16 fbt[NT];
#pragma unroll
    for (int t = 0; t < NT; ++t) {
      fbt[t].h[0] = *reinterpret_cast<const v8bf*>(Bp[t] + k0);
      fbt[t].h[1] = *reinterpret_cast<const v8bf*>(Bp[t] + k0 + 16);
    }
#pragma unroll
    for (int s = 0; s < 2; ++s) {
      const bf16* ap = &sA[buf][(wave * 32 + s * 16 + arow) * 32 + akb];
      Frag16 fa;
      fa.h[0] = *reinterpret_cast<const v8bf*>(ap);
      fa.h[1] = *reinterpret_cast<const v8bf*>(ap + 16);
#pragma unroll
      for (int t = 0; t < NT; ++t)
        acc[s][t] = __builtin_amdgcn_wmma_f32_16x16x32_bf16(
            false, fa.v, false, fbt[t].v, (short)0, acc[s][t], false, false);
    }
    __syncthreads();   // all reads of sA[buf] complete
#if USE_TDM
    if (wave == 0 && more) __builtin_amdgcn_s_wait_tensorcnt(0);
    if (more) __syncthreads();
#else
    if (more) {
      for (int idx = threadIdx.x; idx < 1024; idx += 256) {
        int row = idx >> 2, cc = (idx & 3) << 3;
        *reinterpret_cast<v8bf*>(&sA[buf ^ 1][row * 32 + cc]) =
            *reinterpret_cast<const v8bf*>(A + (size_t)(mBlk + row) * K + (k0 + 32) + cc);
      }
      __syncthreads();
    }
#endif
    buf ^= 1;
  }

  // C/D layout: VGPR r -> M = r (+8 for lanes 16..31), N = lane&15
  const int crow = mBlk + wave * 32 + ((lane >> 4) << 3);
  const int ccol = n0 + (lane & 15);
#pragma unroll
  for (int s = 0; s < 2; ++s)
#pragma unroll
    for (int t = 0; t < NT; ++t)
#pragma unroll
      for (int r = 0; r < 8; ++r)
        C[(size_t)(crow + s * 16 + r) * N + ccol + 16 * t] = acc[s][t][r];
}

// ---------------- dt = softplus(proj_dt + dt_bias) ----------------
__global__ __launch_bounds__(256) void dt_kernel(const float* __restrict__ proj,
                                                 const float* __restrict__ dt_bias,
                                                 float* __restrict__ dtb) {
  int i = blockIdx.x * 256 + threadIdx.x; // over MTOK*NH
  if (i >= MTOK * NH) return;
  int tok = i >> 5, h = i & 31;
  float x = proj[(size_t)tok * PROJ2 + (INTER2 + CONVD) + h] + dt_bias[h];
  dtb[i] = (x > 20.f) ? x : log1pf(__expf(x));
}

// ---------------- depthwise causal conv + bias + SiLU, route streams ----------------
__global__ __launch_bounds__(256) void conv_kernel(const float* __restrict__ proj,
                                                   const float* __restrict__ conv_w,
                                                   const float* __restrict__ conv_b,
                                                   const float* __restrict__ dtb,
                                                   const float* __restrict__ Dp,
                                                   float* __restrict__ hs,
                                                   float* __restrict__ ybuf,
                                                   float* __restrict__ Bm,
                                                   float* __restrict__ Cmo) {
  int i = blockIdx.x * 256 + threadIdx.x; // over MTOK*CONVD
  if (i >= MTOK * CONVD) return;
  int tok = i / CONVD;
  int c = i - tok * CONVD;
  int b = tok >> 11, l = tok & (LSEQ - 1);
  int col = INTER2 + c;
  float s = conv_b[c];
#pragma unroll
  for (int k = 0; k < KCONV; ++k) {
    int li = l - (KCONV - 1) + k;
    if (li >= 0)
      s += conv_w[c * KCONV + k] * proj[(size_t)(b * LSEQ + li) * PROJ2 + col];
  }
  s = s / (1.f + __expf(-s)); // SiLU
  if (c < INTER2) {
    int h = c >> 6;
    hs[(size_t)tok * INTER2 + c]   = s * dtb[tok * NH + h]; // dt-scaled
    ybuf[(size_t)tok * INTER2 + c] = s * Dp[h];             // D residual init
  } else if (c < INTER2 + NDIM) {
    Bm[(size_t)tok * NDIM + (c - INTER2)] = s;
  } else {
    Cmo[(size_t)tok * NDIM + (c - INTER2 - NDIM)] = s;
  }
}

// ---------------- per-(b,h,chunk) cumsum of dt*A ----------------
__global__ __launch_bounds__(256) void acum_kernel(const float* __restrict__ dtb,
                                                   const float* __restrict__ A_log,
                                                   float* __restrict__ acum) {
  int i = blockIdx.x * 256 + threadIdx.x; // over B2*NH*NCH
  if (i >= B2 * NH * NCH) return;
  int c = i % NCH;
  int bh = i / NCH;      // b*NH + h
  int h = bh & 31, b = bh >> 5;
  float a = -__expf(A_log[h]);
  float run = 0.f;
  size_t base = ((size_t)bh * NCH + c) * CHUNK2;
  int tok0 = b * LSEQ + c * CHUNK2;
  for (int l = 0; l < CHUNK2; ++l) {
    run += a * dtb[(size_t)(tok0 + l) * NH + h];
    acum[base + l] = run;
  }
}

// ---------------- Gm[s,z] = sum_n C[s,n]*B[z,n]  (per b,chunk; lower-tri tiles) ----------------
__global__ __launch_bounds__(256) void gm_kernel(const float* __restrict__ Bm,
                                                 const float* __restrict__ Cm,
                                                 float* __restrict__ Gm) {
  int bc = blockIdx.x;              // b*NCH + c
  int st = blockIdx.y, zt = blockIdx.z;
  if (zt > st) return;              // only z<=s needed
  int b = bc / NCH, c = bc % NCH;
  int tokBase = b * LSEQ + c * CHUNK2;
  int s0 = st * 64, z0 = zt * 64;
  int t = threadIdx.x;
  int zz = t & 63, sr = t >> 6;
  __shared__ float sC[64 * 65];
  __shared__ float sB[64 * 65];
  float acc[16];
#pragma unroll
  for (int i = 0; i < 16; ++i) acc[i] = 0.f;
  for (int nt = 0; nt < 2; ++nt) {
    int nb = nt * 64;
    for (int idx = t; idx < 4096; idx += 256) {
      int r = idx >> 6, n = idx & 63;
      sC[r * 65 + n] = Cm[(size_t)(tokBase + s0 + r) * NDIM + nb + n];
      sB[r * 65 + n] = Bm[(size_t)(tokBase + z0 + r) * NDIM + nb + n];
    }
    __syncthreads();
#pragma unroll
    for (int i = 0; i < 16; ++i) {
      int ss = sr + (i << 2);
      float a = 0.f;
      for (int n = 0; n < 64; ++n) a += sC[ss * 65 + n] * sB[zz * 65 + n];
      acc[i] += a;
    }
    __syncthreads();
  }
  float* out = Gm + (size_t)bc * CHUNK2 * CHUNK2;
#pragma unroll
  for (int i = 0; i < 16; ++i)
    out[(size_t)(s0 + sr + (i << 2)) * CHUNK2 + z0 + zz] = acc[i];
}

// ---------------- states[b,c,h,p,n] = sum_l exp(Alast-Acum[l]) * B[l,n] * hs[l,h,p] ----------------
__global__ __launch_bounds__(256) void states_kernel(const float* __restrict__ Bm,
                                                     const float* __restrict__ hs,
                                                     const float* __restrict__ acum,
                                                     float* __restrict__ states) {
  int bch = blockIdx.x;             // (b*NCH + c)*NH + h
  int h = bch % NH;
  int bc = bch / NH;
  int c = bc % NCH, b = bc / NCH;
  int t = threadIdx.x;
  int tokBase = b * LSEQ + c * CHUNK2;
  const float* acumRow = acum + ((size_t)(b * NH + h) * NCH + c) * CHUNK2;
  float alast = acumRow[CHUNK2 - 1];
  __shared__ float sB[64 * 128];    // decayed B tile
  __shared__ float sH[64 * 65];     // hs tile
  float acc[32];
#pragma unroll
  for (int i = 0; i < 32; ++i) acc[i] = 0.f;
  for (int lt = 0; lt < 4; ++lt) {
    int l0 = lt * 64;
    for (int idx = t; idx < 64 * 128; idx += 256) {
      int ll = idx >> 7, nn = idx & 127;
      float dec = __expf(alast - acumRow[l0 + ll]); // <= 1
      sB[idx] = Bm[(size_t)(tokBase + l0 + ll) * NDIM + nn] * dec;
    }
    for (int idx = t; idx < 64 * 64; idx += 256) {
      int ll = idx >> 6, pp = idx & 63;
      sH[ll * 65 + pp] = hs[(size_t)(tokBase + l0 + ll) * INTER2 + h * PDIM + pp];
    }
    __syncthreads();
#pragma unroll
    for (int i = 0; i < 32; ++i) {
      int o = t + 256 * i;         // o = p*128 + n
      int p = o >> 7, n = o & 127;
      float a = 0.f;
      for (int ll = 0; ll < 64; ++ll) a += sB[ll * 128 + n] * sH[ll * 65 + p];
      acc[i] += a;
    }
    __syncthreads();
  }
#pragma unroll
  for (int i = 0; i < 32; ++i)
    states[(size_t)bch * (PDIM * NDIM) + t + 256 * i] = acc[i];
}

// ---------------- inter-chunk serial scan: prev[c] = S ; S = S*exp(T_c) + states[c] ----------------
__global__ __launch_bounds__(256) void scan_kernel(const float* __restrict__ states,
                                                   const float* __restrict__ acum,
                                                   float* __restrict__ prev) {
  int idx = blockIdx.x * 256 + threadIdx.x; // over B2*NH*PDIM*NDIM
  if (idx >= B2 * NH * PDIM * NDIM) return;
  int pn = idx & (PDIM * NDIM - 1);
  int bh = idx >> 13;               // b*NH + h
  int h = bh & 31, b = bh >> 5;
  float S = 0.f;
  for (int c = 0; c < NCH; ++c) {
    size_t sidx = ((size_t)((b * NCH + c) * NH + h)) * (PDIM * NDIM) + pn;
    prev[sidx] = S;
    float alast = acum[((size_t)bh * NCH + c) * CHUNK2 + (CHUNK2 - 1)];
    S = S * __expf(alast) + states[sidx];
  }
}

// ---------------- Y_diag + Y_off, accumulated into y (pre-loaded with D residual) ----------------
__global__ __launch_bounds__(256) void y_kernel(const float* __restrict__ Gm,
                                                const float* __restrict__ acum,
                                                const float* __restrict__ hs,
                                                const float* __restrict__ Cm,
                                                const float* __restrict__ prev,
                                                float* __restrict__ y) {
  int bch = blockIdx.x;             // (b*NCH + c)*NH + h
  int h = bch % NH;
  int bc = bch / NH;
  int c = bc % NCH, b = bc / NCH;
  int st = blockIdx.y;              // s-tile of 64 rows
  int t = threadIdx.x;
  int p = t & 63, sr = t >> 6;
  int tokBase = b * LSEQ + c * CHUNK2;
  int s0 = st * 64;
  const float* acumRow = acum + ((size_t)(b * NH + h) * NCH + c) * CHUNK2;
  const float* GmC = Gm + (size_t)bc * CHUNK2 * CHUNK2;
  const float* prevC = prev + (size_t)bch * (PDIM * NDIM);

  __shared__ float sW[64 * 64];
  __shared__ float sX[64 * 65];
  __shared__ float sAs[64];
  __shared__ float sAz[64];

  float acc[16], acc2[16];
#pragma unroll
  for (int i = 0; i < 16; ++i) { acc[i] = 0.f; acc2[i] = 0.f; }

  if (t < 64) sAs[t] = acumRow[s0 + t];
  __syncthreads();

  // ---- intra-chunk (diagonal blocks), stable masked decay ----
  for (int zt = 0; zt <= st; ++zt) {
    int z0 = zt * 64;
    for (int idx = t; idx < 64 * 64; idx += 256) {
      int zz = idx >> 6, pp = idx & 63;
      sX[zz * 65 + pp] = hs[(size_t)(tokBase + z0 + zz) * INTER2 + h * PDIM + pp];
    }
    if (t < 64) sAz[t] = acumRow[z0 + t];
    __syncthreads();
    for (int idx = t; idx < 64 * 64; idx += 256) {
      int ss = idx >> 6, zz = idx & 63;
      int sA = s0 + ss, zA = z0 + zz;
      float w = 0.f;
      if (zA <= sA)
        w = GmC[(size_t)sA * CHUNK2 + zA] * __expf(sAs[ss] - sAz[zz]); // exp arg <= 0
      sW[idx] = w;
    }
    __syncthreads();
#pragma unroll
    for (int i = 0; i < 16; ++i) {
      int ss = sr + (i << 2);
      float a = 0.f;
      for (int zz = 0; zz < 64; ++zz) a += sW[ss * 64 + zz] * sX[zz * 65 + p];
      acc[i] += a;
    }
    __syncthreads();
  }

  // ---- inter-chunk contribution: sum_n C[s,n]*prev[p,n] ----
  for (int nt = 0; nt < 2; ++nt) {
    int nb = nt * 64;
    for (int idx = t; idx < 64 * 64; idx += 256) {
      int ss = idx >> 6, nn = idx & 63;
      sW[idx] = Cm[(size_t)(tokBase + s0 + ss) * NDIM + nb + nn];
    }
    for (int idx = t; idx < 64 * 64; idx += 256) {
      int pp = idx >> 6, nn = idx & 63;
      sX[pp * 65 + nn] = prevC[(size_t)pp * NDIM + nb + nn];
    }
    __syncthreads();
#pragma unroll
    for (int i = 0; i < 16; ++i) {
      int ss = sr + (i << 2);
      float a = 0.f;
      for (int nn = 0; nn < 64; ++nn) a += sW[ss * 64 + nn] * sX[p * 65 + nn];
      acc2[i] += a;
    }
    __syncthreads();
  }

#pragma unroll
  for (int i = 0; i < 16; ++i) {
    int ss = sr + (i << 2);
    float e = __expf(sAs[ss]); // <= 1
    size_t oi = (size_t)(tokBase + s0 + ss) * INTER2 + h * PDIM + p;
    y[oi] += acc[i] + acc2[i] * e;
  }
}

// ---------------- gated grouped RMSNorm (group = whole INTER, G=1) -> bf16 ----------------
__global__ __launch_bounds__(256) void norm_gate_kernel(const float* __restrict__ y,
                                                        const float* __restrict__ proj,
                                                        const float* __restrict__ norm_w,
                                                        bf16* __restrict__ yn) {
  int tok = blockIdx.x;
  int t = threadIdx.x;
  const float* yrow = y + (size_t)tok * INTER2;
  const float* grow = proj + (size_t)tok * PROJ2; // gate = first INTER cols
  __shared__ float red[256];
  float ss = 0.f;
  for (int i = t; i < INTER2; i += 256) {
    float g = grow[i];
    float v = yrow[i] * (g / (1.f + __expf(-g)));
    ss += v * v;
  }
  red[t] = ss;
  __syncthreads();
  for (int off = 128; off > 0; off >>= 1) {
    if (t < off) red[t] += red[t + off];
    __syncthreads();
  }
  float inv = rsqrtf(red[0] / (float)INTER2 + 1e-6f);
  for (int i = t; i < INTER2; i += 256) {
    float g = grow[i];
    float v = yrow[i] * (g / (1.f + __expf(-g)));
    yn[(size_t)tok * INTER2 + i] = (bf16)(v * inv * norm_w[i]);
  }
}

// ---------------- host launch ----------------
extern "C" void kernel_launch(void* const* d_in, const int* in_sizes, int n_in,
                              void* d_out, int out_size, void* d_ws, size_t ws_size,
                              hipStream_t stream) {
  (void)in_sizes; (void)n_in; (void)out_size;
  const float* x        = (const float*)d_in[0];
  const float* w1       = (const float*)d_in[1]; // [PROJ, DMODEL]
  const float* conv_w   = (const float*)d_in[2];
  const float* conv_b   = (const float*)d_in[3];
  const float* dt_bias  = (const float*)d_in[4];
  const float* A_log    = (const float*)d_in[5];
  const float* Dp       = (const float*)d_in[6];
  const float* norm_w   = (const float*)d_in[7];
  const float* w2       = (const float*)d_in[8]; // [DMODEL, INTER]
  float* out = (float*)d_out;

  char* ws = (char*)d_ws;
  size_t used = 0;
  auto alloc = [&](size_t bytes) -> void* {
    void* p = ws + used;
    used += (bytes + 255) & ~(size_t)255;
    return p;
  };
  bf16*  xbf    = (bf16*)alloc((size_t)MTOK * DMODEL * 2);
  bf16*  w1bf   = (bf16*)alloc((size_t)PROJ2 * DMODEL * 2);
  bf16*  w2bf   = (bf16*)alloc((size_t)DMODEL * INTER2 * 2);
  float* proj   = (float*)alloc((size_t)MTOK * PROJ2 * 4);
  float* dtb    = (float*)alloc((size_t)MTOK * NH * 4);
  float* hs     = (float*)alloc((size_t)MTOK * INTER2 * 4);
  float* ybuf   = (float*)alloc((size_t)MTOK * INTER2 * 4);
  float* Bm     = (float*)alloc((size_t)MTOK * NDIM * 4);
  float* Cm     = (float*)alloc((size_t)MTOK * NDIM * 4);
  float* acum   = (float*)alloc((size_t)B2 * NH * NCH * CHUNK2 * 4);
  float* Gmb    = (float*)alloc((size_t)B2 * NCH * CHUNK2 * CHUNK2 * 4);
  float* states = (float*)alloc((size_t)B2 * NCH * NH * PDIM * NDIM * 4);
  float* prev   = (float*)alloc((size_t)B2 * NCH * NH * PDIM * NDIM * 4);
  bf16*  ynbf   = (bf16*)alloc((size_t)MTOK * INTER2 * 2);
  if (used > ws_size) return; // workspace too small; nothing sensible to do

  // 1) bf16 conversions
  {
    int n;
    n = MTOK * DMODEL;   cvt_bf16_kernel<<<(n + 255) / 256, 256, 0, stream>>>(x,  xbf,  n);
    n = PROJ2 * DMODEL;  cvt_bf16_kernel<<<(n + 255) / 256, 256, 0, stream>>>(w1, w1bf, n);
    n = DMODEL * INTER2; cvt_bf16_kernel<<<(n + 255) / 256, 256, 0, stream>>>(w2, w2bf, n);
  }
  // 2) proj = x @ W1^T   [4096 x 4384] : 68 full 64-col tiles + one 32-col edge
  gemm_bf16_lds<4><<<dim3(MTOK / 256, 68), 256, 0, stream>>>(
      xbf, w1bf, proj, MTOK, PROJ2, DMODEL, 0);
  gemm_bf16_lds<2><<<dim3(MTOK / 256, 1), 256, 0, stream>>>(
      xbf, w1bf, proj, MTOK, PROJ2, DMODEL, 68 * 64);
  // 3) dt softplus
  dt_kernel<<<(MTOK * NH + 255) / 256, 256, 0, stream>>>(proj, dt_bias, dtb);
  // 4) depthwise conv + SiLU + routing (+ D residual into y, dt-scale into hs)
  conv_kernel<<<(MTOK * CONVD + 255) / 256, 256, 0, stream>>>(
      proj, conv_w, conv_b, dtb, Dp, hs, ybuf, Bm, Cm);
  // 5) cumulative A*dt per (b,h,chunk)
  acum_kernel<<<(B2 * NH * NCH + 255) / 256, 256, 0, stream>>>(dtb, A_log, acum);
  // 6) Gm = C * B^T per (b,chunk), lower-tri tiles
  gm_kernel<<<dim3(B2 * NCH, 4, 4), 256, 0, stream>>>(Bm, Cm, Gmb);
  // 7) per-chunk final states
  states_kernel<<<B2 * NCH * NH, 256, 0, stream>>>(Bm, hs, acum, states);
  // 8) inter-chunk recurrence
  scan_kernel<<<(B2 * NH * PDIM * NDIM + 255) / 256, 256, 0, stream>>>(states, acum, prev);
  // 9) Y_diag + Y_off accumulated into y
  y_kernel<<<dim3(B2 * NCH * NH, 4), 256, 0, stream>>>(Gmb, acum, hs, Cm, prev, ybuf);
  // 10) gated RMSNorm -> bf16
  norm_gate_kernel<<<MTOK, 256, 0, stream>>>(ybuf, proj, norm_w, ynbf);
  // 11) out = yn @ W2^T   [4096 x 1024]
  gemm_bf16_lds<4><<<dim3(MTOK / 256, DMODEL / 64), 256, 0, stream>>>(
      ynbf, w2bf, out, MTOK, DMODEL, INTER2, 0);
}
